// MLPResNet_22763326668964
// MI455X (gfx1250) — compile-verified, matrix operand order checked by reading, not run on hardware
//
#include <hip/hip_runtime.h>

typedef __bf16 bf16_t;
typedef __attribute__((ext_vector_type(16))) __bf16 v16bf;
typedef __attribute__((ext_vector_type(8)))  float  v8f;
typedef __attribute__((ext_vector_type(4)))  int    v4i;

#define LDT 72   // padded LDS row (64 bf16 + 8 pad) -> 144B rows, 16B aligned
#define BK  64

// ---- gfx1250 async global->LDS copy (ASYNCcnt-tracked), with safe fallback ----
#if defined(__AMDGCN__) && __has_builtin(__builtin_amdgcn_global_load_async_to_lds_b128)
#define ASYNC_COPY 1
#define AS_GLOBAL(p) ((__attribute__((address_space(1))) v4i*)(p))
#define AS_LOCAL(p)  ((__attribute__((address_space(3))) v4i*)(p))
#else
#define ASYNC_COPY 0
#endif

__device__ __forceinline__ void wait_async0() {
#if ASYNC_COPY
#if __has_builtin(__builtin_amdgcn_s_wait_asynccnt)
  __builtin_amdgcn_s_wait_asynccnt(0);
#else
  asm volatile("s_wait_asynccnt 0x0" ::: "memory");
#endif
#endif
}

__device__ __forceinline__ void copy16(bf16_t* lds, const bf16_t* gsrc) {
#if ASYNC_COPY
  __builtin_amdgcn_global_load_async_to_lds_b128(AS_GLOBAL(gsrc), AS_LOCAL(lds), 0, 0);
#else
  *(uint4*)lds = *(const uint4*)gsrc;
#endif
}

// Fill one 128x64 bf16 A tile and one 128x64 bf16 B^T tile (both row-major,
// contiguous K) with 256 threads: 4 x 16B chunks each per tile.
__device__ __forceinline__ void fill_tiles(bf16_t* as, bf16_t* bs,
                                           const bf16_t* Ablk, const bf16_t* Bblk,
                                           int K, int k0, int frow, int fcol) {
#pragma unroll
  for (int i = 0; i < 4; ++i) {
    const int r = frow + i * 32;
    copy16(as + r * LDT + fcol, Ablk + (size_t)r * K + k0 + fcol);
    copy16(bs + r * LDT + fcol, Bblk + (size_t)r * K + k0 + fcol);
  }
}

// Per-lane WMMA 16-bit fragment load (A 16x32 and B^T share this layout):
// lanes 0-15 hold K=[0..7] and K=[16..23]; lanes 16-31 hold K=[8..15] and K=[24..31]
__device__ __forceinline__ v16bf load_frag(const bf16_t* rowp, int lane) {
  union { uint4 q[2]; v16bf v; } u;
  const int ko = (lane >> 4) * 8;
  u.q[0] = *(const uint4*)(rowp + ko);
  u.q[1] = *(const uint4*)(rowp + 16 + ko);
  return u.v;
}

// out = f(A(bf16, MxK) * Bt(bf16, NxK)^T + bias)
// MODE 0: out = silu(.)   MODE 1: out += silu(.) in-place   MODE 2: out = .
template<int MODE>
__global__ __launch_bounds__(256)
void gemm_mlp_kernel(const bf16_t* __restrict__ A, const bf16_t* __restrict__ Bt,
                     const float* __restrict__ bias, float* out,
                     int M, int N, int K)
{
  __shared__ bf16_t As[2][128 * LDT];
  __shared__ bf16_t Bs[2][128 * LDT];

  const int t      = threadIdx.x;
  const int lane   = t & 31;
  const int wave   = t >> 5;
  const int waveM  = wave & 1;       // 2 waves along M (64 rows each)
  const int waveN  = wave >> 1;      // 4 waves along N (32 cols each)
  const int blockM = blockIdx.y * 128;
  const int blockN = blockIdx.x * 128;

  const int frow = t >> 3;           // loader row group (0..31)
  const int fcol = (t & 7) * 8;      // 8-bf16 chunk within BK=64

  const bf16_t* Ablk = A  + (size_t)blockM * K;
  const bf16_t* Bblk = Bt + (size_t)blockN * K;

  v8f acc[4][2] = {};

  const int nk = K / BK;
  fill_tiles(As[0], Bs[0], Ablk, Bblk, K, 0, frow, fcol);   // prologue stage

  for (int kt = 0; kt < nk; ++kt) {
    const int cur = kt & 1;
    wait_async0();
    __syncthreads();               // stage boundary: consumed buffer landed,
                                   // previous compute done before refill below
    if (kt + 1 < nk)
      fill_tiles(As[cur ^ 1], Bs[cur ^ 1], Ablk, Bblk, K, (kt + 1) * BK, frow, fcol);

    const bf16_t* as = As[cur];
    const bf16_t* bs = Bs[cur];
#pragma unroll
    for (int ks = 0; ks < 2; ++ks) {
      v16bf af[4], bf[2];
#pragma unroll
      for (int tm = 0; tm < 4; ++tm) {
        const int r = waveM * 64 + tm * 16 + (lane & 15);
        af[tm] = load_frag(&as[r * LDT + ks * 32], lane);
      }
#pragma unroll
      for (int tn = 0; tn < 2; ++tn) {
        const int n = waveN * 32 + tn * 16 + (lane & 15);
        bf[tn] = load_frag(&bs[n * LDT + ks * 32], lane);
      }
#pragma unroll
      for (int tm = 0; tm < 4; ++tm)
#pragma unroll
        for (int tn = 0; tn < 2; ++tn)
          acc[tm][tn] = __builtin_amdgcn_wmma_f32_16x16x32_bf16(
              false, af[tm], false, bf[tn], (short)0, acc[tm][tn], false, false);
    }
  }

  // ---- epilogue: bias + SiLU + optional residual ----
  // C/D layout: vgpr r, lanes 0-15 -> M=r, lanes 16-31 -> M=r+8; N = lane&15
  const int rowBase = blockM + waveM * 64 + ((lane >> 4) << 3);
  const int colBase = blockN + waveN * 32 + (lane & 15);
#pragma unroll
  for (int tm = 0; tm < 4; ++tm) {
#pragma unroll
    for (int tn = 0; tn < 2; ++tn) {
      const int col = colBase + tn * 16;
      const float bv = bias[col];
#pragma unroll
      for (int r = 0; r < 8; ++r) {
        const int row = rowBase + tm * 16 + r;
        const size_t idx = (size_t)row * N + col;
        float v = acc[tm][tn][r] + bv;
        if (MODE != 2) v = v / (1.f + __expf(-v));   // SiLU
        if (MODE == 1) v += out[idx];                // residual (in-place)
        out[idx] = v;
      }
    }
  }
}

// W (KxN fp32, row-major) -> Wt (NxK bf16, row-major) via LDS tile transpose
__global__ __launch_bounds__(256)
void transpose_convert_kernel(const float* __restrict__ W, bf16_t* __restrict__ Wt,
                              int K, int N)
{
  __shared__ float tile[32][33];
  const int t  = threadIdx.x;
  const int tx = t & 31;
  const int ty = t >> 5;            // 0..7
  const int kb = blockIdx.y * 32;
  const int nb = blockIdx.x * 32;
#pragma unroll
  for (int i = 0; i < 4; ++i)
    tile[ty + 8 * i][tx] = W[(size_t)(kb + ty + 8 * i) * N + nb + tx];
  __syncthreads();
#pragma unroll
  for (int i = 0; i < 4; ++i)
    Wt[(size_t)(nb + ty + 8 * i) * K + kb + tx] = (bf16_t)tile[tx][ty + 8 * i];
}

// LayerNorm over last dim D (fp32 in) -> bf16 normalized output
__global__ __launch_bounds__(256)
void ln_bf16_kernel(const float* __restrict__ x, const float* __restrict__ scale,
                    const float* __restrict__ bias, bf16_t* __restrict__ out, int D)
{
  const int row = blockIdx.x;
  const float* xr = x + (size_t)row * D;
  bf16_t* orow = out + (size_t)row * D;
  const int t = threadIdx.x;
  const int nvec = D >> 2;

  float s = 0.f, ss = 0.f;
  for (int i = t; i < nvec; i += 256) {
    const float4 v = ((const float4*)xr)[i];
    s  += v.x + v.y + v.z + v.w;
    ss += v.x * v.x + v.y * v.y + v.z * v.z + v.w * v.w;
  }
  __shared__ float rs[256], rss[256];
  rs[t] = s; rss[t] = ss;
  __syncthreads();
  for (int off = 128; off > 0; off >>= 1) {
    if (t < off) { rs[t] += rs[t + off]; rss[t] += rss[t + off]; }
    __syncthreads();
  }
  const float invD = 1.f / (float)D;
  const float mu   = rs[0] * invD;
  const float var  = rss[0] * invD - mu * mu;
  const float rsig = rsqrtf(var + 1e-6f);

  for (int i = t; i < nvec; i += 256) {
    const float4 v  = ((const float4*)xr)[i];
    const float4 sc = ((const float4*)scale)[i];
    const float4 bi = ((const float4*)bias)[i];
    const int c = i * 4;
    orow[c + 0] = (bf16_t)((v.x - mu) * rsig * sc.x + bi.x);
    orow[c + 1] = (bf16_t)((v.y - mu) * rsig * sc.y + bi.y);
    orow[c + 2] = (bf16_t)((v.z - mu) * rsig * sc.z + bi.z);
    orow[c + 3] = (bf16_t)((v.w - mu) * rsig * sc.w + bi.w);
  }
}

extern "C" void kernel_launch(void* const* d_in, const int* in_sizes, int n_in,
                              void* d_out, int out_size, void* d_ws, size_t ws_size,
                              hipStream_t stream) {
  const float* x     = (const float*)d_in[0];
  const float* ln1_s = (const float*)d_in[1];
  const float* ln1_b = (const float*)d_in[2];
  const float* w1    = (const float*)d_in[3];
  const float* b1    = (const float*)d_in[4];
  const float* bls   = (const float*)d_in[5];
  const float* blb   = (const float*)d_in[6];
  const float* bw    = (const float*)d_in[7];
  const float* bb    = (const float*)d_in[8];
  const float* ln2_s = (const float*)d_in[9];
  const float* ln2_b = (const float*)d_in[10];
  const float* w2    = (const float*)d_in[11];
  const float* b2    = (const float*)d_in[12];
  float* out = (float*)d_out;

  const int Mrows = 4 * 2048;   // 8192 tokens
  const int Din = 1024, Dh = 2048, Dout = 1024, NBLK = 6;

  // workspace: [bf16 acts 32MB][f32 hidden 64MB][bf16 W^T 8MB]
  bf16_t* act = (bf16_t*)d_ws;
  float*  h   = (float*)((char*)d_ws + (size_t)Mrows * Dh * sizeof(bf16_t));
  bf16_t* wt  = (bf16_t*)((char*)h + (size_t)Mrows * Dh * sizeof(float));

  // stem: LN1 -> GEMM(w1)+b1 -> SiLU
  transpose_convert_kernel<<<dim3(Dh / 32, Din / 32), 256, 0, stream>>>(w1, wt, Din, Dh);
  ln_bf16_kernel<<<Mrows, 256, 0, stream>>>(x, ln1_s, ln1_b, act, Din);
  gemm_mlp_kernel<0><<<dim3(Dh / 128, Mrows / 128), 256, 0, stream>>>(
      act, wt, b1, h, Mrows, Dh, Din);

  // residual blocks
  for (int i = 0; i < NBLK; ++i) {
    transpose_convert_kernel<<<dim3(Dh / 32, Dh / 32), 256, 0, stream>>>(
        bw + (size_t)i * Dh * Dh, wt, Dh, Dh);
    ln_bf16_kernel<<<Mrows, 256, 0, stream>>>(h, bls + (size_t)i * Dh,
                                              blb + (size_t)i * Dh, act, Dh);
    gemm_mlp_kernel<1><<<dim3(Dh / 128, Mrows / 128), 256, 0, stream>>>(
        act, wt, bb + (size_t)i * Dh, h, Mrows, Dh, Dh);
  }

  // head: LN2 -> GEMM(w2)+b2
  transpose_convert_kernel<<<dim3(Dout / 32, Dh / 32), 256, 0, stream>>>(w2, wt, Dh, Dout);
  ln_bf16_kernel<<<Mrows, 256, 0, stream>>>(h, ln2_s, ln2_b, act, Dh);
  gemm_mlp_kernel<2><<<dim3(Dout / 128, Mrows / 128), 256, 0, stream>>>(
      act, wt, b2, out, Mrows, Dout, Dh);
}
